// RNN_84267258347978
// MI455X (gfx1250) — compile-verified, matrix operand order checked by reading, not run on hardware
//
#include <hip/hip_runtime.h>

#define B_ 8
#define T_ 512
#define E_ 1024
#define H_ 1024
#define V_ 32000

// GEMM tiling
#define BM 64
#define BN 128
#define BK 64
#define LDP 72   // padded LDS row stride in elements (144 B = 9 banks-of-16B)

typedef __attribute__((ext_vector_type(16))) __bf16 v16bf;
typedef __attribute__((ext_vector_type(8)))  float  v8f;
typedef __attribute__((ext_vector_type(4)))  int    v4i;

struct __align__(16) U4 { unsigned int x, y, z, w; };
union FragU { v16bf v; U4 u[2]; };

// Build a 16x32 bf16 fragment from two contiguous 16-byte chunks at p and p+32B.
// Lane r=lane&15 -> matrix row r; lanes 0-15 hold K [k0..k0+7],[k0+16..k0+23];
// lanes 16-31 hold K [k0+8..k0+15],[k0+24..k0+31].
template <typename PT>
__device__ __forceinline__ v16bf frag_from(PT p) {
    FragU f;
    f.u[0] = *(const U4*)p;
    f.u[1] = *(const U4*)(p + 16);
    return f.v;
}

__device__ __forceinline__ v16bf load_frag(const unsigned short* __restrict__ base,
                                           int ld, int row, int k0, int hi) {
    return frag_from(base + (size_t)row * ld + k0 + (hi ? 8 : 0));
}

__device__ __forceinline__ unsigned short f2bf(float f) {
    unsigned int u = __builtin_bit_cast(unsigned int, f);
    u += 0x7FFFu + ((u >> 16) & 1u);          // round-to-nearest-even
    return (unsigned short)(u >> 16);
}

// ----------------------------------------------------- async copy primitives
__device__ __forceinline__ void async_b128(const unsigned short* g, unsigned short* l) {
#if __has_builtin(__builtin_amdgcn_global_load_async_to_lds_b128)
    __builtin_amdgcn_global_load_async_to_lds_b128(
        (__attribute__((address_space(1))) v4i*)g,
        (__attribute__((address_space(3))) v4i*)l, 0, 0);
#else
    unsigned loff = (unsigned)(unsigned long long)
        (__attribute__((address_space(3))) unsigned short*)l;
    asm volatile("global_load_async_to_lds_b128 %0, %1, off"
                 :: "v"(loff), "v"(g) : "memory");
#endif
}

__device__ __forceinline__ void wait_async0() {
#if __has_builtin(__builtin_amdgcn_s_wait_asynccnt)
    __builtin_amdgcn_s_wait_asynccnt(0);
#else
    asm volatile("s_wait_asynccnt 0x0" ::: "memory");
#endif
}

// ---------------------------------------------------------------- conversions
__global__ void cvt_f32_bf16(const float* __restrict__ in,
                             unsigned short* __restrict__ out, int n) {
    int i = blockIdx.x * blockDim.x + threadIdx.x;
    int stride = gridDim.x * blockDim.x;
    for (; i < n; i += stride) out[i] = f2bf(in[i]);
}

__global__ void zero_u16(unsigned short* __restrict__ p, int n) {
    int i = blockIdx.x * blockDim.x + threadIdx.x;
    if (i < n) p[i] = 0;
}

// ----------------------------------------------------- embedding + layernorm
__global__ void embed_ln(const int* __restrict__ x, const float* __restrict__ emb,
                         unsigned short* __restrict__ out) {
    int row = blockIdx.x;                 // token index b*T + t
    int tok = x[row];
    const float* e = emb + (size_t)tok * E_;
    int tid = threadIdx.x;                // 256 threads, 4 elems each
    float v[4];
    float s = 0.f, sq = 0.f;
#pragma unroll
    for (int j = 0; j < 4; j++) {
        float f = e[tid * 4 + j];
        v[j] = f; s += f; sq += f * f;
    }
    __shared__ float sh1[256], sh2[256];
    sh1[tid] = s; sh2[tid] = sq;
    __syncthreads();
    for (int w = 128; w > 0; w >>= 1) {
        if (tid < w) { sh1[tid] += sh1[tid + w]; sh2[tid] += sh2[tid + w]; }
        __syncthreads();
    }
    float mu  = sh1[0] * (1.f / E_);
    float var = sh2[0] * (1.f / E_) - mu * mu;
    float rs  = rsqrtf(var + 1e-5f);
    unsigned short* o = out + (size_t)row * E_;
#pragma unroll
    for (int j = 0; j < 4; j++) o[tid * 4 + j] = f2bf((v[j] - mu) * rs);
}

// ------------------------------------------------------------ WMMA bf16 GEMM
// C[M,N] = act(A[M,K] @ Bw[N,K]^T + bias)   (Bw row-major = B^T)
// Block: 256 threads = 8 waves as 2x4 (M x N); wave tile 32x32; block 64x128.
// K staged through LDS in 64-wide tiles with double-buffered async copies:
//   wait(buf cur) -> barrier -> issue(buf next) (overlaps compute) ->
//   WMMA from ds_load_b128 -> barrier.
template<int OUT_BF16, int ACT>
__global__ void gemm_bf16(const unsigned short* __restrict__ A,
                          const unsigned short* __restrict__ Bw,
                          const float* __restrict__ bias,
                          void* __restrict__ Cout,
                          int M, int N, int K) {
    __shared__ __align__(16) unsigned short sA[2][BM * LDP];
    __shared__ __align__(16) unsigned short sB[2][BN * LDP];

    int tid  = threadIdx.x;
    int lane = tid & 31;
    int wid  = tid >> 5;
    int wave_m = wid >> 2;                 // 0..1
    int wave_n = wid & 3;                  // 0..3
    int mBase = blockIdx.y * BM;
    int nBase = blockIdx.x * BN;
    int r = lane & 15, hi = lane >> 4, o = hi ? 8 : 0;

    // per-thread async copy of one (A,B) k-tile into LDS buffer `buf`
    auto copy_tile = [&](int buf, int kBase) {
#pragma unroll
        for (int j = 0; j < 2; ++j) {                 // A: 512 x 16B chunks
            int c = tid * 2 + j;
            int row = c >> 3, col = (c & 7) * 8;
            async_b128(A + (size_t)(mBase + row) * K + kBase + col,
                       &sA[buf][row * LDP + col]);
        }
#pragma unroll
        for (int j = 0; j < 4; ++j) {                 // B: 1024 x 16B chunks
            int c = tid * 4 + j;
            int row = c >> 3, col = (c & 7) * 8;
            async_b128(Bw + (size_t)(nBase + row) * K + kBase + col,
                       &sB[buf][row * LDP + col]);
        }
    };

    v8f acc[2][2] = {};
    int KT = K / BK;
    copy_tile(0, 0);
    for (int kt = 0; kt < KT; ++kt) {
        int buf = kt & 1;
        wait_async0();
        __syncthreads();                    // buf ready for everyone
        if (kt + 1 < KT) copy_tile(buf ^ 1, (kt + 1) * BK);  // overlap next
#pragma unroll
        for (int kk = 0; kk < BK; kk += 32) {
            v16bf a0 = frag_from(&sA[buf][(wave_m * 32 + r)      * LDP + kk + o]);
            v16bf a1 = frag_from(&sA[buf][(wave_m * 32 + 16 + r) * LDP + kk + o]);
            v16bf b0 = frag_from(&sB[buf][(wave_n * 32 + r)      * LDP + kk + o]);
            v16bf b1 = frag_from(&sB[buf][(wave_n * 32 + 16 + r) * LDP + kk + o]);
            acc[0][0] = __builtin_amdgcn_wmma_f32_16x16x32_bf16(false, a0, false, b0, (short)0, acc[0][0], false, false);
            acc[0][1] = __builtin_amdgcn_wmma_f32_16x16x32_bf16(false, a0, false, b1, (short)0, acc[0][1], false, false);
            acc[1][0] = __builtin_amdgcn_wmma_f32_16x16x32_bf16(false, a1, false, b0, (short)0, acc[1][0], false, false);
            acc[1][1] = __builtin_amdgcn_wmma_f32_16x16x32_bf16(false, a1, false, b1, (short)0, acc[1][1], false, false);
        }
        __syncthreads();                    // done reading buf before refill
    }

#pragma unroll
    for (int mi = 0; mi < 2; mi++) {
#pragma unroll
        for (int ni = 0; ni < 2; ni++) {
            int col = nBase + wave_n * 32 + ni * 16 + r;
            float bv = bias ? bias[col] : 0.f;
#pragma unroll
            for (int v = 0; v < 8; v++) {
                int row = mBase + wave_m * 32 + mi * 16 + v + (hi << 3);
                float f = acc[mi][ni][v] + bv;
                if (ACT == 1) f = fmaxf(f, 0.f);
                if (OUT_BF16)
                    ((unsigned short*)Cout)[(size_t)row * N + col] = f2bf(f);
                else
                    ((float*)Cout)[(size_t)row * N + col] = f;
            }
        }
    }
}

// -------------------------------------------------------------- RNN scan step
// h_t = tanh(xp[:,t,:] + h_{t-1} @ Whh^T + bhh)
// M=8 batch rows padded to 16 (pad rows live only in the [16,H] ping-pong buf).
// Latency-bound serial chain: direct global loads (weights L2-resident).
__global__ void rnn_step(const float* __restrict__ xp,
                         const unsigned short* __restrict__ Whh,
                         const float* __restrict__ bhh,
                         const unsigned short* __restrict__ hprev,
                         unsigned short* __restrict__ hnext,
                         unsigned short* __restrict__ hseq, int t) {
    int lane = threadIdx.x & 31;
    int wid  = threadIdx.x >> 5;          // 0..3
    int n0 = blockIdx.x * 64 + wid * 16;
    int r = lane & 15, hi = lane >> 4;

    v8f acc = {};
    for (int k0 = 0; k0 < H_; k0 += 32) {
        v16bf a = load_frag(hprev, H_, r,      k0, hi);
        v16bf b = load_frag(Whh,   H_, n0 + r, k0, hi);
        acc = __builtin_amdgcn_wmma_f32_16x16x32_bf16(false, a, false, b, (short)0, acc, false, false);
    }
    int col = n0 + r;
    float bv = bhh[col];
#pragma unroll
    for (int v = 0; v < 8; v++) {
        int m = v + (hi << 3);            // batch index 0..15 (8..15 = pad)
        float f = acc[v] + bv;
        if (m < B_) f += xp[((size_t)m * T_ + t) * H_ + col];
        float h = tanhf(f);
        unsigned short hb = f2bf(h);
        hnext[m * H_ + col] = hb;
        if (m < B_) hseq[((size_t)m * T_ + t) * H_ + col] = hb;
    }
}

// ------------------------------------------------------------------ launcher
extern "C" void kernel_launch(void* const* d_in, const int* in_sizes, int n_in,
                              void* d_out, int out_size, void* d_ws, size_t ws_size,
                              hipStream_t stream) {
    (void)in_sizes; (void)n_in; (void)out_size; (void)ws_size;
    const int*   x     = (const int*)d_in[0];
    const float* emb   = (const float*)d_in[1];
    const float* W_ih0 = (const float*)d_in[2];
    const float* W_hh0 = (const float*)d_in[3];
    const float* b_ih0 = (const float*)d_in[4];
    const float* b_hh0 = (const float*)d_in[5];
    const float* W_ih1 = (const float*)d_in[6];
    const float* W_hh1 = (const float*)d_in[7];
    const float* b_ih1 = (const float*)d_in[8];
    const float* b_hh1 = (const float*)d_in[9];
    const float* W_lin = (const float*)d_in[10];
    const float* b_lin = (const float*)d_in[11];

    char* ws = (char*)d_ws;
    size_t off = 0;
    auto alloc = [&](size_t bytes) -> void* {
        void* p = ws + off;
        off += (bytes + 255) & ~(size_t)255;
        return p;
    };
    unsigned short* emb_bf  = (unsigned short*)alloc((size_t)V_ * E_ * 2); // 64 MB, L2-resident
    unsigned short* wih0_bf = (unsigned short*)alloc((size_t)H_ * E_ * 2);
    unsigned short* whh0_bf = (unsigned short*)alloc((size_t)H_ * H_ * 2);
    unsigned short* wih1_bf = (unsigned short*)alloc((size_t)H_ * H_ * 2);
    unsigned short* whh1_bf = (unsigned short*)alloc((size_t)H_ * H_ * 2);
    unsigned short* wlin_bf = (unsigned short*)alloc((size_t)E_ * H_ * 2);
    unsigned short* act_bf  = (unsigned short*)alloc((size_t)B_ * T_ * E_ * 2);
    float*          xp      = (float*)alloc((size_t)B_ * T_ * H_ * 4);
    unsigned short* hseq    = (unsigned short*)alloc((size_t)B_ * T_ * H_ * 2);
    unsigned short* lin_bf  = (unsigned short*)alloc((size_t)B_ * T_ * E_ * 2);
    unsigned short* hping   = (unsigned short*)alloc((size_t)2 * 16 * H_ * 2);

    // f32 -> bf16 operand conversion (embedding + all weight matrices)
    cvt_f32_bf16<<<4096, 256, 0, stream>>>(emb,   emb_bf,  V_ * E_);
    cvt_f32_bf16<<<512,  256, 0, stream>>>(W_ih0, wih0_bf, H_ * E_);
    cvt_f32_bf16<<<512,  256, 0, stream>>>(W_hh0, whh0_bf, H_ * H_);
    cvt_f32_bf16<<<512,  256, 0, stream>>>(W_ih1, wih1_bf, H_ * H_);
    cvt_f32_bf16<<<512,  256, 0, stream>>>(W_hh1, whh1_bf, H_ * H_);
    cvt_f32_bf16<<<512,  256, 0, stream>>>(W_lin, wlin_bf, E_ * H_);

    // gather + layernorm
    embed_ln<<<B_ * T_, 256, 0, stream>>>(x, emb, act_bf);

    dim3 g1(H_ / BN, (B_ * T_) / BM);   // (8, 64)

    // layer 0: input projection, then sequential scan
    gemm_bf16<0, 0><<<g1, 256, 0, stream>>>(act_bf, wih0_bf, b_ih0, xp,
                                            B_ * T_, H_, E_);
    zero_u16<<<(2 * 16 * H_) / 256, 256, 0, stream>>>(hping, 2 * 16 * H_);
    for (int t = 0; t < T_; t++) {
        unsigned short* hp = hping + (t & 1) * 16 * H_;
        unsigned short* hn = hping + ((t + 1) & 1) * 16 * H_;
        rnn_step<<<16, 128, 0, stream>>>(xp, whh0_bf, b_hh0, hp, hn, hseq, t);
    }

    // layer 1
    gemm_bf16<0, 0><<<g1, 256, 0, stream>>>(hseq, wih1_bf, b_ih1, xp,
                                            B_ * T_, H_, H_);
    zero_u16<<<(2 * 16 * H_) / 256, 256, 0, stream>>>(hping, 2 * 16 * H_);
    for (int t = 0; t < T_; t++) {
        unsigned short* hp = hping + (t & 1) * 16 * H_;
        unsigned short* hn = hping + ((t + 1) & 1) * 16 * H_;
        rnn_step<<<16, 128, 0, stream>>>(xp, whh1_bf, b_hh1, hp, hn, hseq, t);
    }

    // linear + relu (bf16 out feeds logits GEMM)
    gemm_bf16<1, 1><<<g1, 256, 0, stream>>>(hseq, wlin_bf, b_lin, lin_bf,
                                            B_ * T_, E_, H_);

    // tied-embedding logits: dominant GEMM (4096 x 32000 x 1024), f32 out
    dim3 g2(V_ / BN, (B_ * T_) / BM);   // (250, 64)
    gemm_bf16<0, 0><<<g2, 256, 0, stream>>>(lin_bf, emb_bf, nullptr, d_out,
                                            B_ * T_, V_, E_);
}
